// MultiHeadAttention_85718957293604
// MI455X (gfx1250) — matrix operand order repaired and checked by reading, not compiled, not run on hardware
//
#include <hip/hip_runtime.h>
#include <hip/hip_bf16.h>

// ---------------------------------------------------------------------------
// MultiHeadAttention + RoPE for MI455X (gfx1250, wave32, WMMA).
// bf16 operands, f32 accumulation via v_wmma_f32_16x16x32_bf16.
// Double-buffered LDS staging via GLOBAL_LOAD_ASYNC_TO_LDS_B128 when the
// toolchain exposes the builtin (ASYNCcnt + s_wait_asynccnt), else sync copy.
// ---------------------------------------------------------------------------

typedef __attribute__((ext_vector_type(16))) __bf16 v16bf;
typedef __attribute__((ext_vector_type(8)))  float  v8f;
typedef int v4i_ __attribute__((vector_size(4 * sizeof(int))));

#define N_ROWS (32 * 512)   // B*T = 16384
#define CDIM   1024
#define HEADS  16
#define HD     64
#define TSEQ   512
#define BSZ    32

#if defined(__has_builtin)
#if __has_builtin(__builtin_amdgcn_global_load_async_to_lds_b128) && \
    __has_builtin(__builtin_amdgcn_s_wait_asynccnt)
#define USE_ASYNC_LDS 1
#endif
#endif
#ifndef USE_ASYNC_LDS
#define USE_ASYNC_LDS 0
#endif

#define AS1 __attribute__((address_space(1)))
#define AS3 __attribute__((address_space(3)))

// 16B global -> LDS copy (per lane). Async path uses ASYNCcnt.
__device__ __forceinline__ void copy16_g2s(__bf16* dst, const __bf16* src) {
#if USE_ASYNC_LDS
    __builtin_amdgcn_global_load_async_to_lds_b128(
        (AS1 v4i_*)(v4i_*)(void*)const_cast<__bf16*>(src),
        (AS3 v4i_*)(v4i_*)(void*)dst, 0, 0);
#else
    *(uint4*)dst = *(const uint4*)src;
#endif
}

__device__ __forceinline__ void async_join() {
#if USE_ASYNC_LDS
    __builtin_amdgcn_s_wait_asynccnt(0);
#endif
}

__device__ __forceinline__ v8f v8f_zero() {
    v8f z;
#pragma unroll
    for (int i = 0; i < 8; ++i) z[i] = 0.0f;
    return z;
}

// --------------------------------------------------------------------------
// fp32 -> bf16 convert
// --------------------------------------------------------------------------
__global__ void cvt_bf16_kernel(const float* __restrict__ s,
                                __bf16* __restrict__ d, int n) {
    int i = blockIdx.x * 256 + threadIdx.x;
    if (i < n) d[i] = (__bf16)s[i];
}

// --------------------------------------------------------------------------
// Y[row, col] = sum_k A[row,k] * W[col,k] + bias[col]
// A: [Nrows x 1024] bf16 row-major, W: [1024 x 1024] bf16 row-major.
// Block = 256 threads (8 waves), 128x128 output tile, K staged in 32-chunks
// through double-buffered LDS. Wave w owns rows [w*16, w*16+16) and all 128
// columns (8 accumulators).
// Fragment layouts used here (A/B in contiguous K order, not the ISA table order):
//   A (16x32 bf16): lane lr=row; hi=lane>>4; elems e -> K=hi*16+e
//   B (32x16 bf16): lane lr=col; elems e -> K=hi*16+e
//   C (16x16 f32):  VGPR i, lane -> M=hi*8+i, N=lr
// --------------------------------------------------------------------------
__global__ __launch_bounds__(256)
void gemm_bias_kernel(const __bf16* __restrict__ A,
                      const __bf16* __restrict__ W,
                      const float* __restrict__ bias,
                      float* __restrict__ Y) {
    __shared__ __align__(16) __bf16 As[2][128 * 40];
    __shared__ __align__(16) __bf16 Bs[2][128 * 40];

    const int tid  = threadIdx.x;
    const int lane = tid & 31;
    const int w    = tid >> 5;
    const int lr   = lane & 15;
    const int hi   = lane >> 4;
    const int rowBase = blockIdx.y * 128;
    const int colBase = blockIdx.x * 128;

    v8f acc[8];
#pragma unroll
    for (int nt = 0; nt < 8; ++nt) acc[nt] = v8f_zero();

    // Stage one 128x32 tile of A and of W into buffer `buf`.
    auto stage = [&](int buf, int k0) {
        for (int c = tid; c < 512; c += 256) {
            const int r  = c >> 2;
            const int qq = c & 3;
            copy16_g2s(&As[buf][r * 40 + qq * 8],
                       A + (size_t)(rowBase + r) * CDIM + k0 + qq * 8);
            copy16_g2s(&Bs[buf][r * 40 + qq * 8],
                       W + (size_t)(colBase + r) * CDIM + k0 + qq * 8);
        }
    };

    stage(0, 0);
    async_join();
    __syncthreads();

    for (int k0 = 0; k0 < CDIM; k0 += 32) {
        const int cur = (k0 >> 5) & 1;
        // Prefetch two K-steps ahead (global_prefetch_b8).
        if (k0 + 64 < CDIM) {
            const int pr = tid >> 1;
            if ((tid & 1) == 0)
                __builtin_prefetch(A + (size_t)(rowBase + pr) * CDIM + k0 + 64,
                                   0, 1);
            else
                __builtin_prefetch(W + (size_t)(colBase + pr) * CDIM + k0 + 64,
                                   0, 1);
        }
        // Kick off async staging of the next K-step into the other buffer.
        if (k0 + 32 < CDIM) stage(cur ^ 1, k0 + 32);

        // Load all fragments first so the 8 WMMAs can issue back-to-back.
        v16bf a;
        const __bf16* ap = &As[cur][(w * 16 + lr) * 40];
#pragma unroll
        for (int e = 0; e < 8; ++e) {
            a[e]     = ap[hi * 16 + e];
            a[8 + e] = ap[hi * 16 + 8 + e];
        }
        v16bf bfr[8];
#pragma unroll
        for (int nt = 0; nt < 8; ++nt) {
            const __bf16* bp = &Bs[cur][(nt * 16 + lr) * 40];
#pragma unroll
            for (int e = 0; e < 16; ++e) bfr[nt][e] = bp[hi * 16 + e];
        }
#pragma unroll
        for (int nt = 0; nt < 8; ++nt) {
            acc[nt] = __builtin_amdgcn_wmma_f32_16x16x32_bf16(
                false, a, false, bfr[nt], (short)0, acc[nt], false, false);
        }

        async_join();     // our async stores into buf^1 are done
        __syncthreads();  // everyone done reading cur / writing nxt
    }

#pragma unroll
    for (int nt = 0; nt < 8; ++nt) {
        const int col = colBase + nt * 16 + lr;
        const float bv = bias[col];
#pragma unroll
        for (int i = 0; i < 8; ++i) {
            const int row = rowBase + w * 16 + hi * 8 + i;
            Y[(size_t)row * CDIM + col] = acc[nt][i] + bv;
        }
    }
}

// --------------------------------------------------------------------------
// RoPE rotate q,k (fp32 in [B,T,H*hd]) and repack q,k,v to bf16 [B,H,T,hd].
// out[i]      = even_i*cos - odd_i*sin   (i in 0..31)
// out[32 + i] = odd_i*cos + even_i*sin
// One thread per (b,h,t,i), i in [0,32).
// --------------------------------------------------------------------------
__global__ void rope_pack_kernel(const float* __restrict__ q,
                                 const float* __restrict__ k,
                                 const float* __restrict__ v,
                                 __bf16* __restrict__ qr,
                                 __bf16* __restrict__ kr,
                                 __bf16* __restrict__ vb) {
    const int idx = blockIdx.x * 256 + threadIdx.x;  // B*H*T*32 total
    const int i = idx & 31;
    const int t = (idx >> 5) & (TSEQ - 1);
    const int h = (idx >> 14) & (HEADS - 1);
    const int b = idx >> 18;
    if (b >= BSZ) return;

    const size_t src = ((size_t)(b * TSEQ + t)) * CDIM + h * HD;
    const size_t dst = ((size_t)((b * HEADS + h) * TSEQ + t)) * HD;

    // theta_i = 10000^(-i/32) = exp(-i * ln(10000)/32)
    const float theta = __expf(-(float)i * 0.2878231366242558f);
    const float ang = (float)t * theta;
    const float s = __sinf(ang);
    const float c = __cosf(ang);

    const float qe = q[src + 2 * i], qo = q[src + 2 * i + 1];
    qr[dst + i]      = (__bf16)(qe * c - qo * s);
    qr[dst + 32 + i] = (__bf16)(qo * c + qe * s);

    const float ke = k[src + 2 * i], ko = k[src + 2 * i + 1];
    kr[dst + i]      = (__bf16)(ke * c - ko * s);
    kr[dst + 32 + i] = (__bf16)(ko * c + ke * s);

    vb[dst + 2 * i]     = (__bf16)v[src + 2 * i];
    vb[dst + 2 * i + 1] = (__bf16)v[src + 2 * i + 1];
}

// --------------------------------------------------------------------------
// Flash attention. Q,K,V in bf16 [B,H,T,hd]. Out bf16 [B,T,H*hd] (= [N,C]).
// Block = 128 threads (4 waves); grid = (T/64, B*H).
// Each wave owns a 16-query strip; K/V streamed 32 keys at a time via LDS.
// Online softmax; P converted C-layout -> A-layout through per-wave LDS.
// --------------------------------------------------------------------------
__global__ __launch_bounds__(128)
void flash_attn_kernel(const __bf16* __restrict__ Q,
                       const __bf16* __restrict__ K,
                       const __bf16* __restrict__ V,
                       __bf16* __restrict__ Out) {
    __shared__ __align__(16) __bf16 Qs[64 * 72];
    __shared__ __align__(16) __bf16 Ks[32 * 72];
    __shared__ __align__(16) __bf16 Vs[32 * 72];
    __shared__ __align__(16) __bf16 Ps[4 * 16 * 40];

    const int tid  = threadIdx.x;
    const int lane = tid & 31;
    const int w    = tid >> 5;
    const int lr   = lane & 15;
    const int hi   = lane >> 4;
    const int bh = blockIdx.y;          // b*H + h
    const int b = bh >> 4, h = bh & 15;
    const int t0 = blockIdx.x * 64;

    const __bf16* Qg = Q + ((size_t)bh * TSEQ + t0) * HD;
    const __bf16* Kg = K + (size_t)bh * TSEQ * HD;
    const __bf16* Vg = V + (size_t)bh * TSEQ * HD;

    // Stage 64x64 Q tile.
    for (int c = tid; c < 512; c += 128) {
        const int r = c >> 3, qq = c & 7;
        copy16_g2s(Qs + r * 72 + qq * 8, Qg + r * 64 + qq * 8);
    }
    async_join();
    __syncthreads();

    // Q A-fragments for K-dim chunks [0,32) and [32,64).
    v16bf aq0, aq1;
    {
        const __bf16* qp = Qs + (w * 16 + lr) * 72;
#pragma unroll
        for (int e = 0; e < 8; ++e) {
            aq0[e]     = qp[hi * 16 + e];
            aq0[8 + e] = qp[hi * 16 + 8 + e];
            aq1[e]     = qp[32 + hi * 16 + e];
            aq1[8 + e] = qp[40 + hi * 16 + e];
        }
    }

    float m_i[8], l_i[8];
    v8f O[4];
#pragma unroll
    for (int i = 0; i < 8; ++i) { m_i[i] = -1e30f; l_i[i] = 0.0f; }
#pragma unroll
    for (int nt = 0; nt < 4; ++nt) O[nt] = v8f_zero();

    const float scale = 0.125f;  // 1/sqrt(64)
    __bf16* Pw = Ps + w * 16 * 40;

    for (int j = 0; j < TSEQ / 32; ++j) {
        __syncthreads();
        // Stage 32x64 K and V tiles (256 16B chunks each).
        for (int c = tid; c < 512; c += 128) {
            const int cc = c & 255;
            const int r = cc >> 3, qq = cc & 7;
            const __bf16* src = (c < 256)
                ? (Kg + ((size_t)(j * 32 + r)) * 64 + qq * 8)
                : (Vg + ((size_t)(j * 32 + r)) * 64 + qq * 8);
            __bf16* dst = (c < 256) ? (Ks + r * 72 + qq * 8)
                                    : (Vs + r * 72 + qq * 8);
            copy16_g2s(dst, src);
        }
        async_join();
        __syncthreads();

        // S[M=query, N=key] over two 16-key sub-tiles.
        v8f S[2];
#pragma unroll
        for (int kt = 0; kt < 2; ++kt) {
            v16bf bk0, bk1;
            const __bf16* kp = Ks + (kt * 16 + lr) * 72;
#pragma unroll
            for (int e = 0; e < 16; ++e) {
                bk0[e] = kp[hi * 16 + e];
                bk1[e] = kp[32 + hi * 16 + e];
            }
            v8f c = v8f_zero();
            c = __builtin_amdgcn_wmma_f32_16x16x32_bf16(
                false, aq0, false, bk0, (short)0, c, false, false);
            c = __builtin_amdgcn_wmma_f32_16x16x32_bf16(
                false, aq1, false, bk1, (short)0, c, false, false);
#pragma unroll
            for (int i = 0; i < 8; ++i) c[i] *= scale;
            S[kt] = c;
        }

        // Online softmax: row M = hi*8 + i lives within one 16-lane half.
#pragma unroll
        for (int i = 0; i < 8; ++i) {
            float mx = fmaxf(S[0][i], S[1][i]);
#pragma unroll
            for (int off = 1; off < 16; off <<= 1)
                mx = fmaxf(mx, __shfl_xor(mx, off, 16));
            const float mn = fmaxf(m_i[i], mx);
            const float alpha = __expf(m_i[i] - mn);
            const float p0 = __expf(S[0][i] - mn);
            const float p1 = __expf(S[1][i] - mn);
            float rs = p0 + p1;
#pragma unroll
            for (int off = 1; off < 16; off <<= 1)
                rs += __shfl_xor(rs, off, 16);
            l_i[i] = l_i[i] * alpha + rs;
            m_i[i] = mn;
            S[0][i] = p0;
            S[1][i] = p1;
            O[0][i] *= alpha; O[1][i] *= alpha;
            O[2][i] *= alpha; O[3][i] *= alpha;
        }

        // C-layout P -> LDS -> A-layout bf16 fragment (16x32).
#pragma unroll
        for (int i = 0; i < 8; ++i) {
            Pw[(hi * 8 + i) * 40 + lr]      = (__bf16)S[0][i];
            Pw[(hi * 8 + i) * 40 + 16 + lr] = (__bf16)S[1][i];
        }
        __syncthreads();

        v16bf apf;
        const __bf16* pp = Pw + lr * 40;
#pragma unroll
        for (int e = 0; e < 8; ++e) {
            apf[e]     = pp[hi * 16 + e];
            apf[8 + e] = pp[hi * 16 + 8 + e];
        }
#pragma unroll
        for (int nt = 0; nt < 4; ++nt) {
            v16bf bv;
#pragma unroll
            for (int e = 0; e < 16; ++e)
                bv[e] = Vs[(hi * 16 + e) * 72 + nt * 16 + lr];
            O[nt] = __builtin_amdgcn_wmma_f32_16x16x32_bf16(
                false, apf, false, bv, (short)0, O[nt], false, false);
        }
    }

    // Normalize and write bf16 [N, C] (row = b*T + t, col = h*64 + d).
    float inv[8];
#pragma unroll
    for (int i = 0; i < 8; ++i) inv[i] = 1.0f / l_i[i];
#pragma unroll
    for (int nt = 0; nt < 4; ++nt) {
#pragma unroll
        for (int i = 0; i < 8; ++i) {
            const int t = t0 + w * 16 + hi * 8 + i;
            const int col = h * HD + nt * 16 + lr;
            Out[((size_t)(b * TSEQ + t)) * CDIM + col] =
                (__bf16)(O[nt][i] * inv[i]);
        }
    }
}

// --------------------------------------------------------------------------
extern "C" void kernel_launch(void* const* d_in, const int* in_sizes, int n_in,
                              void* d_out, int out_size, void* d_ws,
                              size_t ws_size, hipStream_t stream) {
    (void)in_sizes; (void)n_in; (void)out_size; (void)ws_size;

    const float* x  = (const float*)d_in[0];
    const float* Wq = (const float*)d_in[1];
    const float* bq = (const float*)d_in[2];
    const float* Wk = (const float*)d_in[3];
    const float* bk = (const float*)d_in[4];
    const float* Wv = (const float*)d_in[5];
    const float* bv = (const float*)d_in[6];
    const float* Wo = (const float*)d_in[7];
    const float* bo = (const float*)d_in[8];

    char* p = (char*)d_ws;
    auto alloc = [&](size_t bytes) -> void* {
        void* r = (void*)p;
        p += (bytes + 255) & ~(size_t)255;
        return r;
    };

    const size_t NC = (size_t)N_ROWS * CDIM;
    __bf16* xb  = (__bf16*)alloc(NC * 2);
    __bf16* Wqb = (__bf16*)alloc((size_t)CDIM * CDIM * 2);
    __bf16* Wkb = (__bf16*)alloc((size_t)CDIM * CDIM * 2);
    __bf16* Wvb = (__bf16*)alloc((size_t)CDIM * CDIM * 2);
    __bf16* Wob = (__bf16*)alloc((size_t)CDIM * CDIM * 2);
    float*  qf  = (float*)alloc(NC * 4);
    float*  kf  = (float*)alloc(NC * 4);
    float*  vf  = (float*)alloc(NC * 4);
    __bf16* qr  = (__bf16*)alloc(NC * 2);
    __bf16* kr  = (__bf16*)alloc(NC * 2);
    __bf16* vbf = (__bf16*)alloc(NC * 2);
    __bf16* attn = xb;  // x_bf16 is dead after the V projection; reuse.

    // 1. Converts.
    cvt_bf16_kernel<<<(int)((NC + 255) / 256), 256, 0, stream>>>(x, xb, (int)NC);
    const int wn = CDIM * CDIM;
    cvt_bf16_kernel<<<(wn + 255) / 256, 256, 0, stream>>>(Wq, Wqb, wn);
    cvt_bf16_kernel<<<(wn + 255) / 256, 256, 0, stream>>>(Wk, Wkb, wn);
    cvt_bf16_kernel<<<(wn + 255) / 256, 256, 0, stream>>>(Wv, Wvb, wn);
    cvt_bf16_kernel<<<(wn + 255) / 256, 256, 0, stream>>>(Wo, Wob, wn);

    // 2. QKV projections (WMMA, 128x128 tiles).
    dim3 gg(CDIM / 128, N_ROWS / 128);  // (8, 128)
    gemm_bias_kernel<<<gg, 256, 0, stream>>>(xb, Wqb, bq, qf);
    gemm_bias_kernel<<<gg, 256, 0, stream>>>(xb, Wkb, bk, kf);
    gemm_bias_kernel<<<gg, 256, 0, stream>>>(xb, Wvb, bv, vf);

    // 3. RoPE + repack to [B,H,T,hd] bf16.
    const int rope_threads = BSZ * HEADS * TSEQ * 32;  // 8.4M
    rope_pack_kernel<<<rope_threads / 256, 256, 0, stream>>>(qf, kf, vf,
                                                             qr, kr, vbf);

    // 4. Flash attention (WMMA).
    flash_attn_kernel<<<dim3(TSEQ / 64, BSZ * HEADS), 128, 0, stream>>>(
        qr, kr, vbf, attn);

    // 5. Output projection (WMMA) -> fp32 d_out.
    gemm_bias_kernel<<<gg, 256, 0, stream>>>(attn, Wob, bo, (float*)d_out);
}
